// GNNTrans_20452634263694
// MI455X (gfx1250) — matrix operand order repaired and checked by reading, not compiled, not run on hardware
//
#include <hip/hip_runtime.h>
#include <hip/hip_bf16.h>

// ---------------------------------------------------------------------------
// Types for CDNA5 WMMA (wave32): D(16x16 f32) = A(16x32 bf16) x B(32x16 bf16) + C
// ---------------------------------------------------------------------------
typedef __bf16 bf16_t;
typedef bf16_t v16bf __attribute__((ext_vector_type(16)));
typedef float  v8f   __attribute__((ext_vector_type(8)));

#define D_H   128
#define N_OUT 512   // q|k|v|s concatenated

// ---- feature probes --------------------------------------------------------
#if defined(__has_builtin)
#  if __has_builtin(__builtin_amdgcn_global_load_async_to_lds_b128)
#    define HAVE_ASYNC_LDS 1
#  else
#    define HAVE_ASYNC_LDS 0
#  endif
#  if __has_builtin(__builtin_amdgcn_s_wait_asynccnt)
#    define HAVE_WAIT_ASYNC 1
#  else
#    define HAVE_WAIT_ASYNC 0
#  endif
#else
#  define HAVE_ASYNC_LDS 0
#  define HAVE_WAIT_ASYNC 0
#endif

#if HAVE_ASYNC_LDS
#  if HAVE_WAIT_ASYNC
#    define WAIT_ASYNC(N) __builtin_amdgcn_s_wait_asynccnt(N)
#  else
#    define WAIT_ASYNC(N) asm volatile("s_wait_asynccnt " #N ::: "memory")
#  endif
#else
#  define WAIT_ASYNC(N)
#endif

// Pointer types matching the async builtin's prototype (16B int vector).
typedef int b128_t __attribute__((vector_size(16)));
typedef __attribute__((address_space(1))) b128_t* gas_b128;
typedef __attribute__((address_space(3))) b128_t* las_b128;

// Pack two fp32 -> one dword of two bf16 (round-to-nearest-even).
__device__ __forceinline__ unsigned int pk2(float a, float b) {
  unsigned int ua = __float_as_uint(a), ub = __float_as_uint(b);
  ua += 0x7FFFu + ((ua >> 16) & 1u);
  ub += 0x7FFFu + ((ub >> 16) & 1u);
  return (ua >> 16) | (ub & 0xFFFF0000u);
}

// One 16B async (or fallback sync) copy global -> LDS.
__device__ __forceinline__ void stage16(const void* g, void* l) {
#if HAVE_ASYNC_LDS
  __builtin_amdgcn_global_load_async_to_lds_b128(
      (gas_b128)(unsigned long long)g,
      (las_b128)(unsigned int)(unsigned long long)l, 0, 0);
#else
  *(uint4*)l = *(const uint4*)g;
#endif
}

// Monotonic float<->uint mapping so atomicMax(u32) == float max (finite values).
__device__ __forceinline__ unsigned int fmap(float x) {
  unsigned int u = __float_as_uint(x);
  return (u & 0x80000000u) ? ~u : (u | 0x80000000u);
}
__device__ __forceinline__ float funmap(unsigned int u) {
  u = (u & 0x80000000u) ? (u & 0x7FFFFFFFu) : ~u;
  return __uint_as_float(u);
}

// ---------------------------------------------------------------------------
// fp32 -> bf16 row-major conversion, 8 elements (one 16B dst chunk) per thread.
// ---------------------------------------------------------------------------
__global__ void k_cvt_bf16(const float* __restrict__ src, unsigned int* __restrict__ dst,
                           int total8) {
  int t = blockIdx.x * blockDim.x + threadIdx.x;
  if (t >= total8) return;
  float4 f0 = *(const float4*)(src + (size_t)t * 8);
  float4 f1 = *(const float4*)(src + (size_t)t * 8 + 4);
  uint4 o;
  o.x = pk2(f0.x, f0.y); o.y = pk2(f0.z, f0.w);
  o.z = pk2(f1.x, f1.y); o.w = pk2(f1.z, f1.w);
  *(uint4*)(dst + (size_t)t * 4) = o;
}

// ---------------------------------------------------------------------------
// Pack weights [K x 128] x4 -> bf16 B-fragments for V_WMMA_F32_16X16X32_BF16.
// fragIdx = (kt*32 + nt)*32 + lane  ->  out dwords [fragIdx*8 .. fragIdx*8+7]
// ---------------------------------------------------------------------------
__global__ void k_pack_w(const float* __restrict__ w0, const float* __restrict__ w1,
                         const float* __restrict__ w2, const float* __restrict__ w3,
                         int Kdim, unsigned int* __restrict__ out) {
  int tid = blockIdx.x * blockDim.x + threadIdx.x;
  int total = (Kdim >> 5) * 32 * 32;          // ksteps * 32 coltiles * 32 lanes
  if (tid >= total) return;
  int lane = tid & 31;
  int nt   = (tid >> 5) & 31;
  int kt   = tid >> 10;
  int col  = nt * 16 + (lane & 15);
  const float* W = (col < 128) ? w0 : (col < 256) ? w1 : (col < 384) ? w2 : w3;
  int cw = col & 127;
  int kbase = kt * 32 + ((lane & 16) ? 8 : 0);
  unsigned int* o = out + (size_t)tid * 8;
#pragma unroll
  for (int i = 0; i < 8; ++i) {
    int kk = kbase + 2 * i + ((i >= 4) ? 8 : 0);
    float f0 = W[(size_t)(kk)     * D_H + cw];
    float f1 = W[(size_t)(kk + 1) * D_H + cw];
    o[i] = pk2(f0, f1);
  }
}

__global__ void k_pack_bias(const float* __restrict__ b0, const float* __restrict__ b1,
                            const float* __restrict__ b2, const float* __restrict__ b3,
                            float* __restrict__ out) {
  int c = blockIdx.x * blockDim.x + threadIdx.x;
  if (c >= N_OUT) return;
  const float* b = (c < 128) ? b0 : (c < 256) ? b1 : (c < 384) ? b2 : b3;
  out[c] = b[c & 127];
}

// ---------------------------------------------------------------------------
// Fused GEMM:  Out[n x 512] = A[n x Kdim](bf16 row-major) * Wcat + bias
// 256 threads = 8 waves; block owns one 16-row tile, wave w owns col tiles
// w*4..w*4+3. Each K=64 slab (16x64 bf16, 2KB) is staged into LDS via async
// global->LDS (double-buffered, ASYNCcnt); waves consume it as ds_load_b128
// fragments (no conversion ALU in the hot loop) -> 8 WMMAs per slab.
// ---------------------------------------------------------------------------
__global__ __launch_bounds__(256) void k_gemm_qkvs(
    const bf16_t* __restrict__ A, const unsigned int* __restrict__ Bp,
    const float* __restrict__ bias, float* __restrict__ Out, int n, int Kdim) {
  __shared__ bf16_t As[2][16 * 64];

  const int tid  = threadIdx.x;
  const int mt   = blockIdx.x;
  const int lane = tid & 31;
  const int wv   = tid >> 5;
  const int r    = lane & 15;
  const int hi   = lane >> 4;
  const int nslabs = Kdim >> 6;       // K=64 per slab (ksteps always even here)

  // staging coords: threads 0..127 move a 16x64 bf16 slab, 16B each
  const int srow = (tid >> 3) & 15;
  const int sch  = (tid & 7) * 8;     // bf16 elements: 0,8,...,56
  int growc = mt * 16 + srow;
  if (growc >= n) growc = n - 1;      // clamp (stores are masked)
  const bf16_t* gsrc = A + (size_t)growc * Kdim + sch;
  bf16_t* l0 = &As[0][srow * 64 + sch];
  bf16_t* l1 = &As[1][srow * 64 + sch];

  const unsigned int* bw = Bp + (size_t)wv * 1024 + (size_t)lane * 8;

  v8f acc0 = {}, acc1 = {}, acc2 = {}, acc3 = {};

  if (tid < 128) stage16(gsrc, l0);

  for (int sl = 0; sl < nslabs; ++sl) {
    if (sl + 1 < nslabs) {
      if (tid < 128) stage16(gsrc + (sl + 1) * 64, (sl & 1) ? l0 : l1);
      WAIT_ASYNC(1);                  // slab(sl) done, slab(sl+1) in flight
    } else {
      WAIT_ASYNC(0);
    }
    __syncthreads();                  // slab(sl) visible to all waves

    // A fragments (ISA 7.12.2 16-bit A layout):
    // lanes 0-15: K {0..7,16..23}; lanes 16-31: K {8..15,24..31}
    const bf16_t* as0 = &As[sl & 1][r * 64 + (hi ? 8 : 0)];
#pragma unroll
    for (int s = 0; s < 2; ++s) {
      union { uint4 q[2]; v16bf v; } af;
      af.q[0] = *(const uint4*)(as0 + s * 32);        // ds_load_b128
      af.q[1] = *(const uint4*)(as0 + s * 32 + 16);   // ds_load_b128

      const int kt = sl * 2 + s;
      const unsigned int* bk = bw + (size_t)kt * 8192;
      __builtin_prefetch(bk + 16384, 0, 1);           // next slab's B tiles
      v16bf b0 = *(const v16bf*)(bk);
      v16bf b1 = *(const v16bf*)(bk + 256);
      v16bf b2 = *(const v16bf*)(bk + 512);
      v16bf b3 = *(const v16bf*)(bk + 768);

      acc0 = __builtin_amdgcn_wmma_f32_16x16x32_bf16(false, af.v, false, b0, (short)0, acc0, false, false);
      acc1 = __builtin_amdgcn_wmma_f32_16x16x32_bf16(false, af.v, false, b1, (short)0, acc1, false, false);
      acc2 = __builtin_amdgcn_wmma_f32_16x16x32_bf16(false, af.v, false, b2, (short)0, acc2, false, false);
      acc3 = __builtin_amdgcn_wmma_f32_16x16x32_bf16(false, af.v, false, b3, (short)0, acc3, false, false);
    }
    __syncthreads();                  // everyone done reading As[sl&1]
  }

  // D layout: lane<16 -> rows 0..7, lane>=16 -> rows 8..15; col = nt*16 + (lane&15)
  const int cl = lane & 15;
  const int rb = hi ? 8 : 0;
  v8f accs[4] = {acc0, acc1, acc2, acc3};
#pragma unroll
  for (int t = 0; t < 4; ++t) {
    int nt   = wv * 4 + t;
    int colg = nt * 16 + cl;
    float bb = bias[colg];
    v8f a = accs[t];
#pragma unroll
    for (int j = 0; j < 8; ++j) {
      int rr = mt * 16 + rb + j;
      if (rr < n) Out[(size_t)rr * N_OUT + colg] = a[j] + bb;
    }
  }
}

// ---------------------------------------------------------------------------
// Attention kernels. H row layout: [ q(0..127) | k(128..255) | v(256..383) | s(384..511) ]
// ---------------------------------------------------------------------------
__global__ void k_edge_score(const float* __restrict__ H, const int* __restrict__ src,
                             const int* __restrict__ dst, float* __restrict__ score, int E) {
  int gid  = blockIdx.x * blockDim.x + threadIdx.x;
  int e    = gid >> 5;
  int lane = gid & 31;
  if (e >= E) return;
  int s = src[e], d = dst[e];
  float4 q = *(const float4*)(H + (size_t)d * N_OUT + lane * 4);
  float4 k = *(const float4*)(H + (size_t)s * N_OUT + 128 + lane * 4);
  float p = q.x * k.x + q.y * k.y + q.z * k.z + q.w * k.w;
#pragma unroll
  for (int m = 16; m >= 1; m >>= 1) p += __shfl_xor(p, m, 32);
  if (lane == 0) score[e] = p * 0.08838834764831845f;   // 1/sqrt(128)
}

__global__ void k_seg_max(const float* __restrict__ score, const int* __restrict__ dst,
                          unsigned int* __restrict__ smax, int E) {
  int e = blockIdx.x * blockDim.x + threadIdx.x;
  if (e >= E) return;
  atomicMax(&smax[dst[e]], fmap(score[e]));
}

__global__ void k_exp_denom(const float* __restrict__ score, const int* __restrict__ dst,
                            const unsigned int* __restrict__ smax, float* __restrict__ ex,
                            float* __restrict__ denom, int E) {
  int e = blockIdx.x * blockDim.x + threadIdx.x;
  if (e >= E) return;
  int d = dst[e];
  float v = __expf(score[e] - funmap(smax[d]));
  ex[e] = v;
  atomicAdd(&denom[d], v);
}

__global__ void k_aggregate(const float* __restrict__ H, const int* __restrict__ src,
                            const int* __restrict__ dst, const float* __restrict__ ex,
                            const float* __restrict__ denom, float* __restrict__ agg, int E) {
  int gid  = blockIdx.x * blockDim.x + threadIdx.x;
  int e    = gid >> 5;
  int lane = gid & 31;
  if (e >= E) return;
  int s = src[e], d = dst[e];
  float alpha = ex[e] / (denom[d] + 1e-16f);
  float4 v = *(const float4*)(H + (size_t)s * N_OUT + 256 + lane * 4);
  float* ag = agg + (size_t)d * D_H + lane * 4;
  atomicAdd(ag + 0, alpha * v.x);
  atomicAdd(ag + 1, alpha * v.y);
  atomicAdd(ag + 2, alpha * v.z);
  atomicAdd(ag + 3, alpha * v.w);
}

// h1(bf16) = relu(agg + skip), 2 channels per thread (skip = cols 384..511 of H)
__global__ void k_skip_relu_bf16(const float* __restrict__ agg, const float* __restrict__ H,
                                 unsigned int* __restrict__ h1b, int n) {
  int t = blockIdx.x * blockDim.x + threadIdx.x;
  if (t >= n * 64) return;
  int node = t >> 6, c2 = (t & 63) * 2;
  const float* Hs = H + (size_t)node * N_OUT + 384;
  const float* ag = agg + (size_t)node * D_H;
  float f0 = fmaxf(ag[c2]     + Hs[c2],     0.f);
  float f1 = fmaxf(ag[c2 + 1] + Hs[c2 + 1], 0.f);
  h1b[t] = pk2(f0, f1);
}

// ---------------------------------------------------------------------------
// MLP head: gather node -> relu(.@mw1+mb1) -> relu(.@mw2+mb2) -> sigmoid(.@mw3+mb3)
// ---------------------------------------------------------------------------
__global__ __launch_bounds__(128) void k_mlp(
    const float* __restrict__ agg, const float* __restrict__ H, const int* __restrict__ idx,
    const float* __restrict__ mw1, const float* __restrict__ mb1,
    const float* __restrict__ mw2, const float* __restrict__ mb2,
    const float* __restrict__ mw3, const float* __restrict__ mb3,
    float* __restrict__ out) {
  __shared__ float h[128], t1[128], t2[64];
  int g = blockIdx.x, c = threadIdx.x;
  int node = idx[g];
  h[c] = fmaxf(agg[(size_t)node * D_H + c] + H[(size_t)node * N_OUT + 384 + c], 0.f);
  __syncthreads();
  float a = mb1[c];
  for (int j = 0; j < 128; ++j) a += h[j] * mw1[j * 128 + c];
  t1[c] = fmaxf(a, 0.f);
  __syncthreads();
  if (c < 64) {
    float b = mb2[c];
    for (int j = 0; j < 128; ++j) b += t1[j] * mw2[j * 64 + c];
    t2[c] = fmaxf(b, 0.f);
  }
  __syncthreads();
  if (c == 0) {
    float o = mb3[0];
    for (int j = 0; j < 64; ++j) o += t2[j] * mw3[j];
    out[g] = 1.f / (1.f + __expf(-o));
  }
}

// ---------------------------------------------------------------------------
extern "C" void kernel_launch(void* const* d_in, const int* in_sizes, int n_in,
                              void* d_out, int out_size, void* d_ws, size_t ws_size,
                              hipStream_t stream) {
  const float* x   = (const float*)d_in[0];
  const int*   ei  = (const int*)d_in[1];
  const int*   idx = (const int*)d_in[2];
  const float *wq0 = (const float*)d_in[3],  *wk0 = (const float*)d_in[4];
  const float *wv0 = (const float*)d_in[5],  *ws0 = (const float*)d_in[6];
  const float *bq0 = (const float*)d_in[7],  *bk0 = (const float*)d_in[8];
  const float *bv0 = (const float*)d_in[9],  *bs0 = (const float*)d_in[10];
  const float *wq1 = (const float*)d_in[11], *wk1 = (const float*)d_in[12];
  const float *wv1 = (const float*)d_in[13], *ws1 = (const float*)d_in[14];
  const float *bq1 = (const float*)d_in[15], *bk1 = (const float*)d_in[16];
  const float *bv1 = (const float*)d_in[17], *bs1 = (const float*)d_in[18];
  const float *mw1 = (const float*)d_in[19], *mb1 = (const float*)d_in[20];
  const float *mw2 = (const float*)d_in[21], *mb2 = (const float*)d_in[22];
  const float *mw3 = (const float*)d_in[23], *mb3 = (const float*)d_in[24];

  const int n  = in_sizes[0] / 1024;    // 50000 nodes, K0 = 1024
  const int E  = in_sizes[1] / 2;       // 600000 edges
  const int G  = in_sizes[2];           // 1000 graphs
  const int K0 = 1024;

  const int* src = ei;
  const int* dst = ei + E;

  // ---- workspace carve-out (H region reused for layer 0 then layer 1) ----
  char* w = (char*)d_ws;
  size_t off = 0;
  auto alloc = [&](size_t bytes) -> void* {
    void* p = w + off;
    off += (bytes + 255) & ~(size_t)255;
    return p;
  };
  float*        H     = (float*)alloc((size_t)n * N_OUT * 4);   // qkvs (L0, then L1)
  unsigned int* xb    = (unsigned int*)alloc((size_t)n * K0 * 2);   // bf16 x
  unsigned int* h1b   = (unsigned int*)alloc((size_t)n * D_H * 2);  // bf16 h1
  float*        agg   = (float*)alloc((size_t)n * D_H * 4);
  float*        score = (float*)alloc((size_t)E * 4);
  float*        ex    = (float*)alloc((size_t)E * 4);
  unsigned int* smax  = (unsigned int*)alloc((size_t)n * 4);
  float*        denom = (float*)alloc((size_t)n * 4);
  unsigned int* Bp0   = (unsigned int*)alloc((size_t)K0 * N_OUT * 2);
  unsigned int* Bp1   = (unsigned int*)alloc((size_t)D_H * N_OUT * 2);
  float*        bias0 = (float*)alloc(N_OUT * 4);
  float*        bias1 = (float*)alloc(N_OUT * 4);

  const int mtiles = (n + 15) / 16;
  const int ewaves = (E * 32 + 255) / 256;   // wave-per-edge kernels
  const int ethr   = (E + 255) / 256;        // thread-per-edge kernels

  // ---- activation conversion + weight/bias packing ----
  const int x8 = n * (K0 / 8);
  k_cvt_bf16<<<(x8 + 255) / 256, 256, 0, stream>>>(x, xb, x8);
  k_pack_w<<<((K0 >> 5) * 1024 + 255) / 256, 256, 0, stream>>>(wq0, wk0, wv0, ws0, K0, Bp0);
  k_pack_w<<<((D_H >> 5) * 1024 + 255) / 256, 256, 0, stream>>>(wq1, wk1, wv1, ws1, D_H, Bp1);
  k_pack_bias<<<2, 256, 0, stream>>>(bq0, bk0, bv0, bs0, bias0);
  k_pack_bias<<<2, 256, 0, stream>>>(bq1, bk1, bv1, bs1, bias1);

  // ================= layer 0 =================
  k_gemm_qkvs<<<mtiles, 256, 0, stream>>>((const bf16_t*)xb, Bp0, bias0, H, n, K0);

  (void)hipMemsetAsync(smax, 0, (size_t)n * 4, stream);
  (void)hipMemsetAsync(denom, 0, (size_t)n * 4, stream);
  (void)hipMemsetAsync(agg, 0, (size_t)n * D_H * 4, stream);

  k_edge_score<<<ewaves, 256, 0, stream>>>(H, src, dst, score, E);
  k_seg_max<<<ethr, 256, 0, stream>>>(score, dst, smax, E);
  k_exp_denom<<<ethr, 256, 0, stream>>>(score, dst, smax, ex, denom, E);
  k_aggregate<<<ewaves, 256, 0, stream>>>(H, src, dst, ex, denom, agg, E);
  k_skip_relu_bf16<<<(n * 64 + 255) / 256, 256, 0, stream>>>(agg, H, h1b, n);

  // ================= layer 1 (H region reused) =================
  k_gemm_qkvs<<<mtiles, 256, 0, stream>>>((const bf16_t*)h1b, Bp1, bias1, H, n, D_H);

  (void)hipMemsetAsync(smax, 0, (size_t)n * 4, stream);
  (void)hipMemsetAsync(denom, 0, (size_t)n * 4, stream);
  (void)hipMemsetAsync(agg, 0, (size_t)n * D_H * 4, stream);

  k_edge_score<<<ewaves, 256, 0, stream>>>(H, src, dst, score, E);
  k_seg_max<<<ethr, 256, 0, stream>>>(score, dst, smax, E);
  k_exp_denom<<<ethr, 256, 0, stream>>>(score, dst, smax, ex, denom, E);
  k_aggregate<<<ewaves, 256, 0, stream>>>(H, src, dst, ex, denom, agg, E);

  // ================= MLP head =================
  k_mlp<<<G, 128, 0, stream>>>(agg, H, idx, mw1, mb1, mw2, mb2, mw3, mb3, (float*)d_out);
}
